// VQAModel_25022479466686
// MI455X (gfx1250) — compile-verified, matrix operand order checked by reading, not run on hardware
//
#include <hip/hip_runtime.h>

// ---------------------------------------------------------------------------
// MI455X (gfx1250) implementation of the VQA pipeline.
//   K1: convert w_dr / w_ih to bf16 (w_ih zero-padded 216->256 rows)
//   K2: generic bf16-WMMA GEMM  C[M,N] = A[M,K](fp32) * B[N,K](bf16)^T + bias
//       LDS double-buffered: async B copies + A fp32->bf16 conversion for
//       chunk k+1 overlap the 8 WMMAs of chunk k (one barrier / iteration).
//       launch 1: xr = x @ w_dr^T + b_dr           (HBM-bound, ~25us roofline)
//       launch 2: xp = xr @ w_ih^T + b_ih (padded) (tiny)
//   K3: GRU recurrence, single workgroup, 14 waves, 3 WMMAs/step/wave,
//       async-LDS double-buffered xp, fused q = hseq @ w_reg^T + b_reg
//   K4: SITP windows + sigmoid/nlm/lm head -> out[48]
// Workspace requirement: ~36.2 MB.
// ---------------------------------------------------------------------------

typedef __attribute__((ext_vector_type(16))) __bf16        v16bf;
typedef __attribute__((ext_vector_type(8)))  float         v8f;
typedef __attribute__((ext_vector_type(8)))  unsigned int  v8u;
typedef __attribute__((ext_vector_type(4)))  int           v4i_t;

#if __has_builtin(__builtin_amdgcn_global_load_async_to_lds_b128) && \
    __has_builtin(__builtin_amdgcn_s_wait_asynccnt)
#define HAVE_ASYNC_LDS 1
#else
#define HAVE_ASYNC_LDS 0
#endif

#if HAVE_ASYNC_LDS
// Builtin signature (from hipcc diagnostic): param0 = v4i addrspace(1)*
// (non-const), param1 = v4i addrspace(3)*, then imm offset, imm cpol.
static __device__ __forceinline__ void async_ld_b128(const void* g, void* l) {
  __builtin_amdgcn_global_load_async_to_lds_b128(
      (__attribute__((address_space(1))) v4i_t*)(void*)g,
      (__attribute__((address_space(3))) v4i_t*)l, 0, 0);
}
#endif

static __device__ __forceinline__ unsigned short f32_to_bf16(float f) {
  unsigned int u = __float_as_uint(f);
  u += 0x7fffu + ((u >> 16) & 1u);           // round-to-nearest-even
  return (unsigned short)(u >> 16);
}
// Packed fp32x2 -> bf16x2: two v_add (round-half-up) + one v_perm_b32 that
// merges the two high halves (D.lo16 = a.hi16, D.hi16 = b.hi16).
static __device__ __forceinline__ unsigned int pack_bf16x2(float a, float b) {
  unsigned int ua = __float_as_uint(a) + 0x8000u;
  unsigned int ub = __float_as_uint(b) + 0x8000u;
  return __builtin_amdgcn_perm(ub, ua, 0x07060302u);
}
// Load one 16x32 bf16 WMMA A/B fragment slice for this lane from LDS.
// Layout per CDNA5 ISA: lanes 0-15 hold K = {0..7, 16..23}, lanes 16-31
// hold K = {8..15, 24..31}; caller passes p = &row[koff] with koff = 0 or 8.
static __device__ __forceinline__ v16bf ld_frag(const unsigned short* p) {
  uint4 lo = *(const uint4*)p;          // K: koff .. koff+7
  uint4 hi = *(const uint4*)(p + 16);   // K: koff+16 .. koff+23
  v8u u = { lo.x, lo.y, lo.z, lo.w, hi.x, hi.y, hi.z, hi.w };
  return __builtin_bit_cast(v16bf, u);
}
static __device__ __forceinline__ float sigmoidf_(float x) {
  return 1.0f / (1.0f + __expf(-x));
}
static __device__ __forceinline__ float fast_tanhf(float x) {
  float xc = fminf(fmaxf(x, -15.0f), 15.0f);
  float e = __expf(2.0f * xc);
  return (e - 1.0f) / (e + 1.0f);
}

// ---------------------------------------------------------------------------
// K1: weight conversion.  w_dr [256,8704] -> bf16 as-is.
//     w_ih [216,256] -> bf16 padded to [256,256] (pad rows zero) so B-staging
//     in the GEMM never needs a bounds guard (required for the async path).
// ---------------------------------------------------------------------------
#define WDR_ELEMS (256 * 8704)
#define WIH_PAD_ELEMS (256 * 256)

__global__ __launch_bounds__(256) void convert_weights(
    const float* __restrict__ w_dr, const float* __restrict__ w_ih,
    unsigned short* __restrict__ wdr_bf, unsigned short* __restrict__ wih_bf) {
  int tid = blockIdx.x * 256 + threadIdx.x;
  if (tid < WDR_ELEMS) {
    wdr_bf[tid] = f32_to_bf16(w_dr[tid]);
  } else {
    int t2 = tid - WDR_ELEMS;
    if (t2 < WIH_PAD_ELEMS) {
      int r = t2 >> 8, c = t2 & 255;
      wih_bf[t2] = (r < 216) ? f32_to_bf16(w_ih[r * 256 + c]) : (unsigned short)0;
    }
  }
}

// ---------------------------------------------------------------------------
// K2: generic bf16 WMMA GEMM.  Workgroup tile 128M x 64N, 8 waves, each wave
// owns a 32x32 block (2x2 of 16x16 accumulators).  K chunked by 64 with LDS
// double buffering: while chunk k is consumed by WMMAs, chunk k+1's B tiles
// are async-copied to LDS and A tiles are loaded/converted/stored; a single
// barrier per iteration separates the ping-pong buffers.
// ---------------------------------------------------------------------------
#define MT 128
#define NT 64
#define KC 64
#define APAD 72   // 64 bf16 + 8 pad -> 144 B rows, conflict-free b128 frags
#define BPAD 72

__global__ __launch_bounds__(256) void gemm_bf16_wmma(
    const float* __restrict__ A, int lda,
    const unsigned short* __restrict__ B, int ldb,
    const float* __restrict__ bias, int bias_len,
    float* __restrict__ C, int ldc, int N, int K) {
  __shared__ unsigned short sA[2][MT * APAD];   // 2 x 18,432 B
  __shared__ unsigned short sB[2][NT * BPAD];   // 2 x  9,216 B
  const int tid = threadIdx.x;
  const int lane = tid & 31, wid = tid >> 5;
  const int m0 = blockIdx.x * MT, n0 = blockIdx.y * NT;
  const int wm = (wid & 3) * 32, wn = (wid >> 2) * 32;
  const int arow = lane & 15;
  const int koff = (lane < 16) ? 0 : 8;

  // Per-thread staging bases, hoisted out of the K loop.
  // A: 8 units, rows ar + 16p, cols ac..ac+3.  B: 2 units, rows br + 32p.
  const int ar = tid >> 4, ac = (tid & 15) * 4;
  const int br = tid >> 3, bc = (tid & 7) * 8;
  const float* aP = A + (size_t)(m0 + ar) * lda + ac;
  const unsigned short* bP = B + (size_t)(n0 + br) * ldb + bc;
  const size_t aStep = (size_t)16 * lda;
  const size_t bStep = (size_t)32 * ldb;
  unsigned short* sAp[2] = { &sA[0][ar * APAD + ac], &sA[1][ar * APAD + ac] };
  unsigned short* sBp[2] = { &sB[0][br * BPAD + bc], &sB[1][br * BPAD + bc] };

  v8f acc[2][2] = {};
  float4 f[8];

  auto issue = [&](int kc, int buf) {
#pragma unroll
    for (int p = 0; p < 2; ++p) {
#if HAVE_ASYNC_LDS
      async_ld_b128(bP + p * bStep + kc, sBp[buf] + p * 32 * BPAD);
#else
      *(uint4*)(sBp[buf] + p * 32 * BPAD) = *(const uint4*)(bP + p * bStep + kc);
#endif
    }
#pragma unroll
    for (int p = 0; p < 8; ++p)
      f[p] = *(const float4*)(aP + p * aStep + kc);
  };
  auto commit = [&](int buf) {
#pragma unroll
    for (int p = 0; p < 8; ++p)
      *(uint2*)(sAp[buf] + p * 16 * APAD) =
          make_uint2(pack_bf16x2(f[p].x, f[p].y), pack_bf16x2(f[p].z, f[p].w));
#if HAVE_ASYNC_LDS
    __builtin_amdgcn_s_wait_asynccnt(0);
#endif
  };

  // Prologue: stage chunk 0 into buffer 0.
  issue(0, 0);
  commit(0);
  __syncthreads();

  for (int kc = 0; kc < K; kc += KC) {
    const int cur = (kc >> 6) & 1;
    const bool notlast = (kc + KC) < K;
    if (notlast) issue(kc + KC, cur ^ 1);

#pragma unroll
    for (int c = 0; c < 2; ++c) {
      v16bf af0 = ld_frag(sA[cur] + (wm + arow) * APAD + c * 32 + koff);
      v16bf af1 = ld_frag(sA[cur] + (wm + 16 + arow) * APAD + c * 32 + koff);
      v16bf bf0 = ld_frag(sB[cur] + (wn + arow) * BPAD + c * 32 + koff);
      v16bf bf1 = ld_frag(sB[cur] + (wn + 16 + arow) * BPAD + c * 32 + koff);
      acc[0][0] = __builtin_amdgcn_wmma_f32_16x16x32_bf16(
          false, af0, false, bf0, (short)0, acc[0][0], false, false);
      acc[0][1] = __builtin_amdgcn_wmma_f32_16x16x32_bf16(
          false, af0, false, bf1, (short)0, acc[0][1], false, false);
      acc[1][0] = __builtin_amdgcn_wmma_f32_16x16x32_bf16(
          false, af1, false, bf0, (short)0, acc[1][0], false, false);
      acc[1][1] = __builtin_amdgcn_wmma_f32_16x16x32_bf16(
          false, af1, false, bf1, (short)0, acc[1][1], false, false);
    }

    if (notlast) commit(cur ^ 1);
    __syncthreads();
  }

  // Epilogue: C/D layout — VGPR v, lanes 0-15: (M=v, N=lane); lanes 16-31:
  // (M=8+v, N=lane-16).
#pragma unroll
  for (int mi = 0; mi < 2; ++mi)
#pragma unroll
    for (int ni = 0; ni < 2; ++ni) {
      int col = n0 + wn + ni * 16 + arow;
      if (col < N) {
        float bi = (col < bias_len) ? bias[col] : 0.0f;
        int rb = m0 + wm + mi * 16 + ((lane < 16) ? 0 : 8);
#pragma unroll
        for (int v = 0; v < 8; ++v)
          C[(size_t)(rb + v) * ldc + col] = acc[mi][ni][v] + bi;
      }
    }
}

// ---------------------------------------------------------------------------
// K3: GRU.  Single workgroup, 448 threads = 14 waves (wave w owns output
// columns [16w,16w+16) of the 16x224 padded hp = h @ w_hh^T).  K=72 padded to
// 96 -> 3 WMMAs per wave per timestep; w_hh B-fragments live in VGPRs across
// the whole t-loop.  xp[t+1] is async-copied to an LDS double buffer during
// step t's WMMA phase (ASYNCcnt-counted wait keeps exactly one copy in
// flight).  Gates + h update + q-dot fused; 2 barriers per step.
// ---------------------------------------------------------------------------
#define GH 72
#define GH3 216
#define NP 224      // padded 3H
#define HPAD 104    // bf16 row pitch for h (96 used + pad, 16B-aligned)
#define WPAD 104

__global__ __launch_bounds__(448) void gru_wmma(
    const float* __restrict__ xp,      // [16*1024][224], includes b_ih
    const float* __restrict__ w_hh,    // [216][72]
    const float* __restrict__ b_hh,    // [216]
    const float* __restrict__ w_reg,   // [72]
    const float* __restrict__ b_reg,   // [1]
    float* __restrict__ q) {           // [16][1024]
  __shared__ unsigned short sW[NP * WPAD];   // w_hh bf16, zero padded
  __shared__ unsigned short sHb[16 * HPAD];  // h bf16 (A-matrix source)
  __shared__ float sHf[16 * GH];             // h fp32
  __shared__ float sHp[16 * NP];             // h @ w_hh^T
  __shared__ float sBhh[NP];
  __shared__ float sWreg[GH];
#if HAVE_ASYNC_LDS
  __shared__ float sXp[2][16 * NP];          // double-buffered xp[t] slice
#endif
  const int tid = threadIdx.x, lane = tid & 31, wid = tid >> 5;

  for (int i = tid; i < NP * WPAD; i += 448) sW[i] = 0;
  for (int i = tid; i < 16 * HPAD; i += 448) sHb[i] = 0;
  for (int i = tid; i < 16 * GH; i += 448) sHf[i] = 0.0f;
  for (int i = tid; i < NP; i += 448) sBhh[i] = (i < GH3) ? b_hh[i] : 0.0f;
  for (int i = tid; i < GH; i += 448) sWreg[i] = w_reg[i];
  __syncthreads();
  for (int i = tid; i < GH3 * GH; i += 448) {
    int r = i / GH, c = i - r * GH;
    sW[r * WPAD + c] = f32_to_bf16(w_hh[i]);
  }
  __syncthreads();

  const int brow = lane & 15;
  const int koff = (lane < 16) ? 0 : 8;
  v16bf Bf[3];                               // resident across all 1024 steps
#pragma unroll
  for (int c = 0; c < 3; ++c)
    Bf[c] = ld_frag(sW + (wid * 16 + brow) * WPAD + c * 32 + koff);

  const float breg = b_reg[0];
  const int col = wid * 16 + brow;
  const int rbase = (lane < 16) ? 0 : 8;

#if HAVE_ASYNC_LDS
  // xp slice = 16 rows x 224 floats = 896 b128 units; 2 per thread.
  int ub[2], uc[2];
#pragma unroll
  for (int k = 0; k < 2; ++k) {
    int u = tid + k * 448;
    ub[k] = u / 56;                   // batch row
    uc[k] = (u - ub[k] * 56) * 4;     // float column
  }
#pragma unroll
  for (int k = 0; k < 2; ++k)
    async_ld_b128(xp + (size_t)(ub[k] * 1024 + 0) * NP + uc[k],
                  &sXp[0][ub[k] * NP + uc[k]]);
  __builtin_amdgcn_s_wait_asynccnt(0);
#endif

  for (int t = 0; t < 1024; ++t) {
#if HAVE_ASYNC_LDS
    if (t + 1 < 1024) {
#pragma unroll
      for (int k = 0; k < 2; ++k)
        async_ld_b128(xp + (size_t)(ub[k] * 1024 + t + 1) * NP + uc[k],
                      &sXp[(t + 1) & 1][ub[k] * NP + uc[k]]);
    }
#endif
    // hp tile = h(bf16) @ w_hh^T for this wave's 16 columns.
    v8f acc = {};
#pragma unroll
    for (int c = 0; c < 3; ++c) {
      v16bf a = ld_frag(sHb + brow * HPAD + c * 32 + koff);
      acc = __builtin_amdgcn_wmma_f32_16x16x32_bf16(
          false, a, false, Bf[c], (short)0, acc, false, false);
    }
#pragma unroll
    for (int v = 0; v < 8; ++v) sHp[(rbase + v) * NP + col] = acc[v];
#if HAVE_ASYNC_LDS
    // Ensure step t's xp copy landed; step t+1's 2 loads may stay in flight.
    if (t + 1 < 1024) __builtin_amdgcn_s_wait_asynccnt(2);
    else              __builtin_amdgcn_s_wait_asynccnt(0);
#endif
    __syncthreads();

    // Gates (torch order r,z,n) + state update, 1152 elements over 448 lanes.
    for (int i = tid; i < 16 * GH; i += 448) {
      int b = i / GH, j = i - b * GH;
#if HAVE_ASYNC_LDS
      const float* xrow = &sXp[t & 1][b * NP];
#else
      const float* xrow = xp + (size_t)(b * 1024 + t) * NP;
#endif
      float r = sigmoidf_(xrow[j] + sHp[b * NP + j] + sBhh[j]);
      float z = sigmoidf_(xrow[GH + j] + sHp[b * NP + GH + j] + sBhh[GH + j]);
      float n = fast_tanhf(xrow[2 * GH + j] +
                           r * (sHp[b * NP + 2 * GH + j] + sBhh[2 * GH + j]));
      float h = (1.0f - z) * n + z * sHf[i];
      sHf[i] = h;
      sHb[b * HPAD + j] =
          (unsigned short)((__float_as_uint(h) + 0x8000u) >> 16);
    }
    __syncthreads();

    // q[b,t] = h . w_reg + b_reg  (deterministic serial reduction per row).
    if (tid < 16) {
      float s = breg;
#pragma unroll 8
      for (int j = 0; j < GH; ++j) s += sHf[tid * GH + j] * sWreg[j];
      q[tid * 1024 + t] = s;
    }
  }
}

// ---------------------------------------------------------------------------
// K4: SITP windows + head.  q fits in LDS (64 KB of 320 KB/WGP).
// out[0:16]=relative, out[16:32]=mapped, out[32:48]=aligned.
// ---------------------------------------------------------------------------
__global__ __launch_bounds__(256) void sitp_head(
    const float* __restrict__ q, const int* __restrict__ x_len,
    const float* __restrict__ nlm_w1, const float* __restrict__ nlm_b1,
    const float* __restrict__ nlm_w2, const float* __restrict__ nlm_b2,
    const float* __restrict__ lm_w, const float* __restrict__ lm_b,
    float* __restrict__ out) {
  __shared__ float sq[16 * 1024];
  for (int i = threadIdx.x; i < 16 * 1024; i += 256) sq[i] = q[i];
  __syncthreads();
  const int lane = threadIdx.x & 31, wid = threadIdx.x >> 5;
  for (int b = wid; b < 16; b += 8) {
    const float* row = sq + b * 1024;
    const int len = x_len[b];         // in [TAU, T]
    float partial = 0.0f;
    for (int t = lane; t < len; t += 32) {
      // trailing min over [t-11, t]; all s <= t < len are valid here
      float mn = __builtin_inff();
      const int s0 = (t >= 11) ? (t - 11) : 0;
      for (int s = s0; s <= t; ++s) mn = fminf(mn, row[s]);
      // forward exp(-q)-weighted mean over [t, t+11] within valid range
      float num = 0.0f, den = 0.0f;
      const int s1 = (t + 11 < len) ? (t + 11) : (len - 1);
      for (int s = t; s <= s1; ++s) {
        float w = __expf(-row[s]);
        num += w * row[s];
        den += w;
      }
      float y = num / fmaxf(den, 1e-30f);
      partial += 0.5f * y + 0.5f * mn;   // GAMMA = 0.5
    }
#pragma unroll
    for (int off = 16; off > 0; off >>= 1)
      partial += __shfl_down(partial, off, 32);
    if (lane == 0) {
      float rel = sigmoidf_(partial / (float)len);
      float mp = sigmoidf_(nlm_w1[0] * rel + nlm_b1[0]) * nlm_w2[0] + nlm_b2[0];
      float al = lm_w[0] * mp + lm_b[0];
      out[b] = rel;
      out[16 + b] = mp;
      out[32 + b] = al;
    }
  }
}

// ---------------------------------------------------------------------------
extern "C" void kernel_launch(void* const* d_in, const int* in_sizes, int n_in,
                              void* d_out, int out_size, void* d_ws,
                              size_t ws_size, hipStream_t stream) {
  (void)in_sizes; (void)n_in; (void)out_size; (void)ws_size;
  const float* x      = (const float*)d_in[0];   // [16,1024,8704]
  const int*   x_len  = (const int*)  d_in[1];   // [16]
  const float* w_dr   = (const float*)d_in[2];   // [256,8704]
  const float* b_dr   = (const float*)d_in[3];   // [256]
  const float* w_ih   = (const float*)d_in[4];   // [216,256]
  const float* w_hh   = (const float*)d_in[5];   // [216,72]
  const float* b_ih   = (const float*)d_in[6];   // [216]
  const float* b_hh   = (const float*)d_in[7];   // [216]
  const float* w_reg  = (const float*)d_in[8];   // [1,72]
  const float* b_reg  = (const float*)d_in[9];   // [1]
  const float* nlm_w1 = (const float*)d_in[10];
  const float* nlm_b1 = (const float*)d_in[11];
  const float* nlm_w2 = (const float*)d_in[12];
  const float* nlm_b2 = (const float*)d_in[13];
  const float* lm_w   = (const float*)d_in[14];
  const float* lm_b   = (const float*)d_in[15];

  // Workspace layout (256B-aligned offsets), total ~36.2 MB.
  char* ws = (char*)d_ws;
  unsigned short* wdr_bf = (unsigned short*)(ws + 0);          // 4,456,448 B
  unsigned short* wih_bf = (unsigned short*)(ws + 4456448);    //   131,072 B
  float* xr = (float*)(ws + 4587520);                          // 16,777,216 B
  float* xp = (float*)(ws + 21364736);                         // 14,680,064 B
  float* qb = (float*)(ws + 36044800);                         //    65,536 B

  // K1: weight conversion (2,293,760 elements).
  convert_weights<<<8960, 256, 0, stream>>>(w_dr, w_ih, wdr_bf, wih_bf);

  // K2a: xr[16384,256] = x @ w_dr^T + b_dr   (K=8704, HBM-bound)
  gemm_bf16_wmma<<<dim3(128, 4), 256, 0, stream>>>(
      x, 8704, wdr_bf, 8704, b_dr, 256, xr, 256, 256, 8704);

  // K2b: xp[16384,224] = xr @ w_ih^T + b_ih  (N padded 216->224, K=256)
  gemm_bf16_wmma<<<dim3(128, 4), 256, 0, stream>>>(
      xr, 256, wih_bf, 256, b_ih, 216, xp, 224, 224, 256);

  // K3: GRU recurrence + q projection (one workgroup carries the recurrence).
  gru_wmma<<<1, 448, 0, stream>>>(xp, w_hh, b_hh, w_reg, b_reg, qb);

  // K4: SITP + head.
  sitp_head<<<1, 256, 0, stream>>>(qb, x_len, nlm_w1, nlm_b1, nlm_w2, nlm_b2,
                                   lm_w, lm_b, (float*)d_out);
}
